// LanguageGaussianRasterizer_84851373900164
// MI455X (gfx1250) — matrix operand order repaired and checked by reading, not compiled, not run on hardware
//
#include <hip/hip_runtime.h>
#include <hip/hip_bf16.h>

typedef __attribute__((ext_vector_type(16))) _Float16 v16h;
typedef __attribute__((ext_vector_type(8)))  float    v8f;

// ---------------- problem constants ----------------
#define NG      1024
#define LFEAT   8
#define HIMG    128
#define WIMG    128
#define NPIX    (HIMG * WIMG)          // 16384
#define TANFOV  0.7f

// output offsets (flat float elements in d_out, in reference return order)
#define OFF_COLOR    0                 // 3*16384
#define OFF_LANG     49152             // 8*16384
#define OFF_RADII    180224            // 1024 (int32 bits)
#define OFF_RADII_L  181248            // 1024
#define OFF_DEPTH    182272            // 16384
#define OFF_OPAC     198656            // 16384
#define OFF_OPACL    215040            // 16384
#define OFF_NT       231424            // 1024 (int32 bits)
#define OFF_NTL      232448            // 1024

// ---------------- kernel 1: per-gaussian preprocess ----------------
// packed params per gaussian: {px, py, ca, cb, cc, op, tz, thr}
// thr = valid ? log(1/(255*op)) : +inf   (alpha>0  <=>  thr <= power <= 0)
__global__ __launch_bounds__(256)
void preprocess_kernel(const float* __restrict__ means3D,
                       const float* __restrict__ opac0,
                       const float* __restrict__ opac1,
                       const float* __restrict__ scales0,
                       const float* __restrict__ scales1,
                       const float* __restrict__ rots0,
                       const float* __restrict__ rots1,
                       const float* __restrict__ V,   // viewmatrix row-major 4x4
                       const float* __restrict__ PM,  // projmatrix row-major 4x4
                       float* __restrict__ params,    // [2][NG][8]
                       float* __restrict__ keys,      // [2][NG]
                       float* __restrict__ out)       // full output buffer
{
    int tid = blockIdx.x * blockDim.x + threadIdx.x;
    if (tid >= 2 * NG) return;
    int r = tid / NG, n = tid % NG;

    // zero n_touched regions (poisoned by harness; count kernel uses atomics)
    ((int*)out)[OFF_NT + tid] = 0;

    const float* sc = r ? scales1 : scales0;
    const float* qt = r ? rots1 : rots0;
    const float* op = r ? opac1 : opac0;

    float m0 = means3D[n * 3 + 0], m1 = means3D[n * 3 + 1], m2 = means3D[n * 3 + 2];

    // t = [m,1] @ V
    float t0 = m0 * V[0] + m1 * V[4] + m2 * V[8]  + V[12];
    float t1 = m0 * V[1] + m1 * V[5] + m2 * V[9]  + V[13];
    float tz = m0 * V[2] + m1 * V[6] + m2 * V[10] + V[14];

    // ph = [m,1] @ PM
    float p0 = m0 * PM[0] + m1 * PM[4] + m2 * PM[8]  + PM[12];
    float p1 = m0 * PM[1] + m1 * PM[5] + m2 * PM[9]  + PM[13];
    float p3 = m0 * PM[3] + m1 * PM[7] + m2 * PM[11] + PM[15];
    float pw = 1.0f / (p3 + 1e-7f);
    float px = ((p0 * pw + 1.0f) * (float)WIMG - 1.0f) * 0.5f;
    float py = ((p1 * pw + 1.0f) * (float)HIMG - 1.0f) * 0.5f;

    // quaternion -> rotation
    float qr = qt[n*4+0], qx = qt[n*4+1], qy = qt[n*4+2], qz = qt[n*4+3];
    float qn = rsqrtf(qr*qr + qx*qx + qy*qy + qz*qz);
    qr *= qn; qx *= qn; qy *= qn; qz *= qn;
    float R00 = 1.f - 2.f*(qy*qy + qz*qz), R01 = 2.f*(qx*qy - qr*qz), R02 = 2.f*(qx*qz + qr*qy);
    float R10 = 2.f*(qx*qy + qr*qz), R11 = 1.f - 2.f*(qx*qx + qz*qz), R12 = 2.f*(qy*qz - qr*qx);
    float R20 = 2.f*(qx*qz - qr*qy), R21 = 2.f*(qy*qz + qr*qx), R22 = 1.f - 2.f*(qx*qx + qy*qy);

    float s0 = sc[n*3+0], s1 = sc[n*3+1], s2 = sc[n*3+2];  // SCALE_MOD = 1
    float M00 = R00*s0, M01 = R01*s1, M02 = R02*s2;
    float M10 = R10*s0, M11 = R11*s1, M12 = R12*s2;
    float M20 = R20*s0, M21 = R21*s1, M22 = R22*s2;

    float C00 = M00*M00 + M01*M01 + M02*M02;
    float C01 = M00*M10 + M01*M11 + M02*M12;
    float C02 = M00*M20 + M01*M21 + M02*M22;
    float C11 = M10*M10 + M11*M11 + M12*M12;
    float C12 = M10*M20 + M11*M21 + M12*M22;
    float C22 = M20*M20 + M21*M21 + M22*M22;

    float fx = 0.5f * (float)WIMG / TANFOV;
    float fy = 0.5f * (float)HIMG / TANFOV;
    float limx = 1.3f * TANFOV, limy = 1.3f * TANFOV;
    float txz = fminf(limx, fmaxf(-limx, t0 / tz)) * tz;
    float tyz = fminf(limy, fmaxf(-limy, t1 / tz)) * tz;
    float itz = 1.0f / tz;
    float j00 = fx * itz, j02 = -fx * txz * itz * itz;
    float j11 = fy * itz, j12 = -fy * tyz * itz * itz;

    float T00 = j00*V[0] + j02*V[2];
    float T01 = j00*V[4] + j02*V[6];
    float T02 = j00*V[8] + j02*V[10];
    float T10 = j11*V[1] + j12*V[2];
    float T11 = j11*V[5] + j12*V[6];
    float T12 = j11*V[9] + j12*V[10];

    float a0 = T00*C00 + T01*C01 + T02*C02;
    float a1 = T00*C01 + T01*C11 + T02*C12;
    float a2 = T00*C02 + T01*C12 + T02*C22;
    float b0 = T10*C00 + T11*C01 + T12*C02;
    float b1 = T10*C01 + T11*C11 + T12*C12;
    float b2 = T10*C02 + T11*C12 + T12*C22;
    float c00 = a0*T00 + a1*T01 + a2*T02 + 0.3f;
    float c01 = a0*T10 + a1*T11 + a2*T12;
    float c11 = b0*T10 + b1*T11 + b2*T12 + 0.3f;

    float det = c00 * c11 - c01 * c01;
    float inv = 1.0f / (det + 1e-12f);
    float ca =  c11 * inv;
    float cb = -c01 * inv;
    float cc =  c00 * inv;
    float mid = 0.5f * (c00 + c11);
    float lam1 = mid + sqrtf(fmaxf(0.1f, mid * mid - det));
    bool valid = (tz > 0.2f) && (det > 0.0f);
    int radii = valid ? (int)ceilf(3.0f * sqrtf(lam1)) : 0;

    ((int*)out)[(r ? OFF_RADII_L : OFF_RADII) + n] = radii;

    float opv = op[n];
    float* pr = params + (size_t)(r * NG + n) * 8;
    pr[0] = px; pr[1] = py; pr[2] = ca; pr[3] = cb; pr[4] = cc;
    pr[5] = opv; pr[6] = tz;
    pr[7] = valid ? logf(1.0f / (255.0f * opv)) : __builtin_inff();
    keys[r * NG + n] = valid ? tz : __builtin_inff();
}

// ---------------- kernel 2: n_touched (scan-independent, exp-free) ----------------
// thread = (render, gaussian, 256-pixel chunk): 2*1024*64 threads
__global__ __launch_bounds__(256)
void ntouched_kernel(const float* __restrict__ params,  // [2][NG][8] (unsorted)
                     float* __restrict__ out)
{
    int tid = blockIdx.x * blockDim.x + threadIdx.x;
    int r     = tid >> 16;            // NG*64 = 65536 threads per render
    int g     = (tid >> 6) & (NG - 1);
    int chunk = tid & 63;

    const float* pr = params + (size_t)(r * NG + g) * 8;
    float gx = pr[0], gy = pr[1], ca = pr[2], cb = pr[3], cc = pr[4], thr = pr[7];

    int base = chunk * 256;
    int cnt = 0;
    #pragma unroll 4
    for (int i = 0; i < 256; ++i) {
        int p = base + i;
        float dx = gx - (float)(p & (WIMG - 1));
        float dy = gy - (float)(p >> 7);
        float power = -0.5f * (ca * dx * dx + cc * dy * dy) - cb * dx * dy;
        cnt += (power <= 0.0f && power >= thr) ? 1 : 0;
    }
    if (cnt)
        atomicAdd(&((int*)out)[(r ? OFF_NTL : OFF_NT) + g], cnt);
}

// ---------------- kernel 3: bitonic argsort + gather + B-matrix build ----------------
__global__ __launch_bounds__(1024)
void sort_gather_kernel(const float* __restrict__ keys,    // [2][NG]
                        const float* __restrict__ params,  // [2][NG][8]
                        const float* __restrict__ colors,  // [NG][3]
                        const float* __restrict__ lang,    // [NG][LFEAT]
                        float* __restrict__ sparams,       // [2][NG][8]
                        _Float16* __restrict__ featsT)     // [2][16][NG]
{
    __shared__ float skey[NG];
    __shared__ int   sidx[NG];
    int r = blockIdx.x;
    int i = threadIdx.x;
    skey[i] = keys[r * NG + i];
    sidx[i] = i;
    __syncthreads();

    #pragma unroll 1
    for (int k = 2; k <= NG; k <<= 1) {
        #pragma unroll 1
        for (int j = k >> 1; j > 0; j >>= 1) {
            int ixj = i ^ j;
            if (ixj > i) {
                bool up = ((i & k) == 0);
                float ka = skey[i], kb = skey[ixj];
                if ((ka > kb) == up) {
                    skey[i] = kb; skey[ixj] = ka;
                    int t = sidx[i]; sidx[i] = sidx[ixj]; sidx[ixj] = t;
                }
            }
            __syncthreads();
        }
    }

    int n = sidx[i];
    const float* src = params  + (size_t)(r * NG + n) * 8;
    float*       dst = sparams + (size_t)(r * NG + i) * 8;
    #pragma unroll
    for (int q = 0; q < 8; ++q) dst[q] = src[q];

    _Float16* ft = featsT + (size_t)r * 16 * NG;
    if (r == 0) {
        ft[0 * NG + i] = (_Float16)colors[n * 3 + 0];
        ft[1 * NG + i] = (_Float16)colors[n * 3 + 1];
        ft[2 * NG + i] = (_Float16)colors[n * 3 + 2];
        ft[3 * NG + i] = (_Float16)src[6];      // depth channel (tz)
        ft[4 * NG + i] = (_Float16)1.0f;        // ones -> 1 - T_final
        #pragma unroll
        for (int c = 5; c < 16; ++c) ft[c * NG + i] = (_Float16)0.0f;
    } else {
        #pragma unroll
        for (int c = 0; c < LFEAT; ++c) ft[c * NG + i] = (_Float16)lang[n * LFEAT + c];
        ft[8 * NG + i] = (_Float16)1.0f;
        #pragma unroll
        for (int c = 9; c < 16; ++c) ft[c * NG + i] = (_Float16)0.0f;
    }
}

// ---------------- async global->LDS staging (gfx1250 async path) ----------------
// every thread copies one dword; tracked by ASYNCcnt
__device__ __forceinline__ void stage_chunk_async(const float* gsrc, float* lbuf, int t) {
    unsigned           lds = (unsigned)(unsigned long long)(uintptr_t)(&lbuf[t]);
    unsigned long long ga  = (unsigned long long)(uintptr_t)(gsrc + t);
    asm volatile("global_load_async_to_lds_b32 %0, %1, off"
                 :: "v"(lds), "v"(ga) : "memory");
}
__device__ __forceinline__ void wait_async0() {
    asm volatile("s_wait_asynccnt 0x0" ::: "memory");
}

// ---------------- kernel 4: rasterize (1 wave = 16-pixel tile, WMMA blend) ----------------
__global__ __launch_bounds__(256)
void raster_kernel(const float* __restrict__ sparams,   // [2][NG][8]
                   const _Float16* __restrict__ featsT, // [2][16][NG]
                   const float* __restrict__ bg,        // [3]
                   float* __restrict__ out)
{
    __shared__ __align__(16) float pbuf[2][256];   // double-buffered 32-gaussian chunks

    const int r    = blockIdx.y;
    const int t    = threadIdx.x;
    const int lane = t & 31;
    const int wave = t >> 5;
    const int tile = blockIdx.x * 8 + wave;      // 0..1023, 16 pixels each
    const int row  = lane & 15;                  // A-row (pixel) / B,D-col (channel)
    const int hi   = lane >> 4;                  // half-wave id

    const int p = tile * 16 + row;
    const float pxf = (float)(p & (WIMG - 1));
    const float pyf = (float)(p >> 7);

    const float*    gp = sparams + (size_t)r * NG * 8;
    const _Float16* bT = featsT + (size_t)r * 16 * NG;

    v8f acc = {};
    float Tchunk = 1.0f;

    // prologue: stage chunk 0
    stage_chunk_async(gp, pbuf[0], t);
    wait_async0();
    __syncthreads();

    for (int c = 0; c < NG / 32; ++c) {
        // stage next chunk while computing on this one
        if (c + 1 < NG / 32)
            stage_chunk_async(gp + (c + 1) * 256, pbuf[(c + 1) & 1], t);

        const int k0 = c * 32;
        // B fragment: lane holds channel row, K = {hi*8..hi*8+7, 16+hi*8..16+hi*8+7}
        const _Float16* bsrc = bT + (size_t)row * NG + k0 + hi * 8;
        v16h bfrag;
        #pragma unroll
        for (int s = 0; s < 8; ++s) { bfrag[s] = bsrc[s]; bfrag[s + 8] = bsrc[16 + s]; }
        if (k0 + 32 < NG)
            __builtin_prefetch(bT + (size_t)row * NG + k0 + 32 + hi * 16, 0, 1);

        const float* lp = pbuf[c & 1];

        // A fragment K layout (16-bit A 16x32): lane-lo K={0..7,16..23}, lane-hi K={8..15,24..31}
        float aarr[16];
        float prodA = 1.0f, prodB = 1.0f;
        #pragma unroll
        for (int seg = 0; seg < 2; ++seg) {
            #pragma unroll
            for (int j = 0; j < 8; ++j) {
                const float* g = lp + (size_t)(seg * 16 + hi * 8 + j) * 8;
                float4 A = *(const float4*)(g);
                float4 B = *(const float4*)(g + 4);
                float dx = A.x - pxf, dy = A.y - pyf;
                float power = -0.5f * (A.z * dx * dx + B.x * dy * dy) - A.w * dx * dy;
                float e = __expf(power);
                bool ok = (power <= 0.0f) && (power >= B.w);
                float alpha = ok ? fminf(0.99f, B.y * e) : 0.0f;
                aarr[seg * 8 + j] = alpha;
                if (seg) prodB *= (1.0f - alpha); else prodA *= (1.0f - alpha);
            }
        }

        // exchange partial (1-alpha) products to split the transmittance scan
        float oA = __shfl_xor(prodA, 16, 32);
        float oB = __shfl_xor(prodB, 16, 32);
        float P0 = hi ? oA : prodA;      // K 0..7
        float P1 = hi ? prodA : oA;      // K 8..15
        float P2 = hi ? oB : prodB;      // K 16..23
        float P3 = hi ? prodB : oB;      // K 24..31

        float Tseg0 = Tchunk * (hi ? P0 : 1.0f);
        float Tseg1 = Tchunk * P0 * P1 * (hi ? P2 : 1.0f);

        v16h afrag;
        float Tl = Tseg0;
        #pragma unroll
        for (int j = 0; j < 8; ++j) {
            afrag[j] = (_Float16)(aarr[j] * Tl);
            Tl *= (1.0f - aarr[j]);
        }
        Tl = Tseg1;
        #pragma unroll
        for (int j = 8; j < 16; ++j) {
            afrag[j] = (_Float16)(aarr[j] * Tl);
            Tl *= (1.0f - aarr[j]);
        }
        Tchunk *= P0 * P1 * P2 * P3;

        acc = __builtin_amdgcn_wmma_f32_16x16x32_f16(
            false, afrag, false, bfrag, (short)0, acc, false, false);

        // next chunk staged + everyone done reading this buffer
        wait_async0();
        __syncthreads();
    }

    // D layout: lane holds channel c = row; VGPR v -> pixel m = tile*16 + v + 8*hi.
    const int ones_ch = r ? 8 : 4;
    #pragma unroll
    for (int v = 0; v < 8; ++v) {
        float sw = __shfl(acc[v], (lane & 16) + ones_ch, 32);  // sum(w) = 1 - T_final
        int m = tile * 16 + v + 8 * hi;
        if (r == 0) {
            if (row < 3)       out[OFF_COLOR + row * NPIX + m] = acc[v] + (1.0f - sw) * bg[row];
            else if (row == 3) out[OFF_DEPTH + m] = acc[v];
            else if (row == 4) out[OFF_OPAC  + m] = sw;
        } else {
            if (row < LFEAT)   out[OFF_LANG + row * NPIX + m] = acc[v];
            else if (row == 8) out[OFF_OPACL + m] = sw;
        }
    }
}

// ---------------- host-side launch ----------------
extern "C" void kernel_launch(void* const* d_in, const int* in_sizes, int n_in,
                              void* d_out, int out_size, void* d_ws, size_t ws_size,
                              hipStream_t stream) {
    const float* means3D = (const float*)d_in[0];
    const float* colors  = (const float*)d_in[2];
    const float* lang    = (const float*)d_in[3];
    const float* opac0   = (const float*)d_in[4];
    const float* opac1   = (const float*)d_in[5];
    const float* scales0 = (const float*)d_in[6];
    const float* scales1 = (const float*)d_in[7];
    const float* rots0   = (const float*)d_in[8];
    const float* rots1   = (const float*)d_in[9];
    const float* viewm   = (const float*)d_in[10];
    const float* projm   = (const float*)d_in[11];
    const float* bg      = (const float*)d_in[12];

    char* ws = (char*)d_ws;
    float*    params  = (float*)(ws + 0);        // 64 KB
    float*    sparams = (float*)(ws + 65536);    // 64 KB
    float*    keys    = (float*)(ws + 131072);   // 8 KB
    _Float16* featsT  = (_Float16*)(ws + 139264);// 64 KB
    float* out = (float*)d_out;

    preprocess_kernel<<<dim3(8), dim3(256), 0, stream>>>(
        means3D, opac0, opac1, scales0, scales1, rots0, rots1,
        viewm, projm, params, keys, out);

    ntouched_kernel<<<dim3(512), dim3(256), 0, stream>>>(params, out);

    sort_gather_kernel<<<dim3(2), dim3(1024), 0, stream>>>(
        keys, params, colors, lang, sparams, featsT);

    raster_kernel<<<dim3(128, 2), dim3(256), 0, stream>>>(
        sparams, featsT, bg, out);
}
